// CustomWaveletLayer_45243185496816
// MI455X (gfx1250) — matrix operand order repaired and checked by reading, not compiled, not run on hardware
//
#include <hip/hip_runtime.h>

#define B_  1024
#define I_  128
#define O_  128
#define W_  8
#define NN  32              // Chebyshev nodes (Lagrange basis size)
#define K_  (I_ * NN)       // GEMM reduction dim = 4096 (128 k-steps of 32)
#define KSTEPS (K_ / 32)    // 128
#define PI_F 3.14159265358979323846f

typedef __attribute__((ext_vector_type(16))) _Float16 v16h;
typedef __attribute__((ext_vector_type(8)))  _Float16 v8h;
typedef __attribute__((ext_vector_type(8)))  float    v8f;

__device__ __forceinline__ float morlet_f(float u) {
    return __cosf(5.0f * u) * __expf(-0.5f * u * u);
}

// ---------------------------------------------------------------------------
// Swizzled fragment layouts (per 16-row tile `t`, per k-step `ks`: 64 chunks
// of 8 halves = 1KB, chunk index c = part*32 + hl*16 + mr):
//   A-swizzle: chunk holds A[t*16+mr, ks*32 + hl*8  + part*16 + (0..7)]
//   H-swizzle: chunk holds H[t*16+mr, ks*32 + hl*8  + part*16 + (0..7)]
// In the GEMM, lane L = hl*16+mr loads chunk L (frag elems 0..7) and chunk
// L+32 (frag elems 8..15): fully coalesced 512B per b128 instruction.
// ---------------------------------------------------------------------------

// Kernel 1: H[o, i*NN+n] = sum_w coef[o,i,w]*morlet(x_n*zoom[o,i,w]-pan[o,i,w])
// One thread per (o,i,n), written directly in swizzled fragment order.
__global__ __launch_bounds__(256) void build_H_kernel(
    const float* __restrict__ coef, const float* __restrict__ zoom,
    const float* __restrict__ pan, _Float16* __restrict__ Hs)
{
    const int id = blockIdx.x * blockDim.x + threadIdx.x;   // = (o*I_+i)*NN + n
    if (id >= O_ * I_ * NN) return;
    const int n  = id & (NN - 1);
    const int oi = id >> 5;                                  // o*I_ + i
    const int o  = oi >> 7;                                  // I_ = 128
    const int i  = oi & (I_ - 1);

    const float xn = __cosf(PI_F * ((float)n + 0.5f) / (float)NN);

    const float* cp = coef + (size_t)oi * W_;
    const float* zp = zoom + (size_t)oi * W_;
    const float* pp = pan  + (size_t)oi * W_;
    float acc = 0.0f;
#pragma unroll
    for (int w = 0; w < W_; ++w) {
        float u = fmaf(xn, zp[w], -pp[w]);
        acc = fmaf(cp[w], morlet_f(u), acc);
    }

    // B-fragment swizzle: ks = i, hl = (n%16)/8, part = n/16, e = n%8
    const int ot    = o >> 4;
    const int mr    = o & 15;
    const int hl    = (n >> 3) & 1;
    const int part  = n >> 4;
    const int e     = n & 7;
    const int chunk = part * 32 + hl * 16 + mr;
    Hs[(((size_t)ot * KSTEPS + i) * 64 + chunk) * 8 + e] = (_Float16)acc;
}

// Kernel 2: A[b, i*NN+n] = L_n(tanh(x[b,i]*r)), barycentric Lagrange at
// Chebyshev nodes; each thread owns one full k-step (ks = i) of one row and
// writes four coalescible 16B chunks in swizzled fragment order.
__global__ __launch_bounds__(256) void build_A_kernel(
    const float* __restrict__ x, const float* __restrict__ tanh_range,
    _Float16* __restrict__ As)
{
    const int id = blockIdx.x * blockDim.x + threadIdx.x;    // = b*I_ + i
    if (id >= B_ * I_) return;
    const int b = id >> 7;                                   // I_ = 128
    const int i = id & (I_ - 1);
    const float t = tanhf(x[id] * tanh_range[0]);

    float rn[NN];
    float s = 0.0f;
#pragma unroll
    for (int n = 0; n < NN; ++n) {
        const float ang = PI_F * (2.0f * (float)n + 1.0f) / (2.0f * (float)NN);
        const float xn  = __cosf(ang);
        const float wn  = ((n & 1) ? -1.0f : 1.0f) * __sinf(ang);
        float d = t - xn;
        d = (fabsf(d) < 1e-12f) ? copysignf(1e-12f, d) : d;  // node-hit guard
        const float v = __fdividef(wn, d);
        rn[n] = v;
        s += v;
    }
    const float inv = __fdividef(1.0f, s);

    _Float16 tv[NN];
#pragma unroll
    for (int n = 0; n < NN; ++n) tv[n] = (_Float16)(rn[n] * inv);

    // A-fragment swizzle: n 0..7 -> chunk mr (hl0,part0); 8..15 -> 16+mr
    // (hl1,part0); 16..23 -> 32+mr (hl0,part1); 24..31 -> 48+mr (hl1,part1).
    const int bt = b >> 4;
    const int mr = b & 15;
    v8h* dst = (v8h*)(As + ((size_t)bt * KSTEPS + i) * 512);  // 64 chunks * 8
    v8h c0, c1, c2, c3;
#pragma unroll
    for (int e = 0; e < 8; ++e) {
        c0[e] = tv[e];      c1[e] = tv[8 + e];
        c2[e] = tv[16 + e]; c3[e] = tv[24 + e];
    }
    dst[mr]      = c0;
    dst[16 + mr] = c1;
    dst[32 + mr] = c2;
    dst[48 + mr] = c3;
}

// ---------------------------------------------------------------------------
// Kernel 3: out[b,o] = sum_k A[b,k]*H[o,k]  (M=1024, N=128, K=4096)
// One wave32 per 16x16 C tile; 512 tiles; 128 v_wmma_f32_16x16x32_f16 each.
// Fragments come from the pre-swizzled buffers: per k-step each operand is a
// contiguous 1KB block; lane L loads chunk L and chunk L+32 (two b128 loads,
// fully coalesced).
// ---------------------------------------------------------------------------
__global__ __launch_bounds__(256) void wmma_gemm_kernel(
    const _Float16* __restrict__ As, const _Float16* __restrict__ Hs,
    float* __restrict__ out)
{
    const int lane = threadIdx.x & 31;
    const int wave = threadIdx.x >> 5;
    const int tile = blockIdx.x * 8 + wave;     // 64 blocks * 8 waves = 512
    const int bt   = tile >> 3;                 // 0..63  (b tile)
    const int ot   = tile & 7;                  // 0..7   (o tile)
    const int hl   = lane >> 4;
    const int mr   = lane & 15;

    const _Float16* abase = As + (size_t)bt * KSTEPS * 512 + (size_t)lane * 8;
    const _Float16* hbase = Hs + (size_t)ot * KSTEPS * 512 + (size_t)lane * 8;

    v8f c = {};
    for (int ks = 0; ks < KSTEPS; ++ks) {
        const _Float16* ap = abase + (size_t)ks * 512;
        const _Float16* hp = hbase + (size_t)ks * 512;
        __builtin_prefetch(ap + 1024, 0, 3);    // 2 k-steps ahead -> prefetch_b8
        __builtin_prefetch(hp + 1024, 0, 3);

        const v8h alo = *(const v8h*)(ap);          // chunk lane    (elems 0..7)
        const v8h ahi = *(const v8h*)(ap + 256);    // chunk lane+32 (elems 8..15)
        const v8h blo = *(const v8h*)(hp);
        const v8h bhi = *(const v8h*)(hp + 256);

        v16h af, bf;
#pragma unroll
        for (int e = 0; e < 8; ++e) {
            af[e] = alo[e]; af[8 + e] = ahi[e];
            bf[e] = blo[e]; bf[8 + e] = bhi[e];
        }
        // (neg_a, A, neg_b, B, c_mod, C, reuse_a, reuse_b)
        c = __builtin_amdgcn_wmma_f32_16x16x32_f16(false, af, false, bf,
                                                   (short)0, c, false, false);
    }

    // C layout: VGPR r -> (M = r + 8*hl, N = mr)
    float* obase = out + (size_t)(bt * 16 + hl * 8) * O_ + (ot * 16 + mr);
#pragma unroll
    for (int r = 0; r < 8; ++r) obase[(size_t)r * O_] = c[r];
}

// ---------------------------------------------------------------------------
extern "C" void kernel_launch(void* const* d_in, const int* in_sizes, int n_in,
                              void* d_out, int out_size, void* d_ws, size_t ws_size,
                              hipStream_t stream)
{
    const float* x    = (const float*)d_in[0];   // (B, I)
    const float* tr   = (const float*)d_in[1];   // scalar
    const float* coef = (const float*)d_in[2];   // (O, I, W)
    const float* zoom = (const float*)d_in[3];   // (O, I, W)
    const float* pan  = (const float*)d_in[4];   // (O, I, W)
    float* out = (float*)d_out;                  // (B, O)

    _Float16* Amat = (_Float16*)d_ws;                                   // 8 MB
    _Float16* Hmat = (_Float16*)((char*)d_ws + (size_t)B_ * K_ * 2);    // 1 MB

    build_H_kernel<<<(O_ * I_ * NN) / 256, 256, 0, stream>>>(coef, zoom, pan, Hmat);
    build_A_kernel<<<(B_ * I_) / 256, 256, 0, stream>>>(x, tr, Amat);
    wmma_gemm_kernel<<<(B_ / 16) * (O_ / 16) / 8, 256, 0, stream>>>(Amat, Hmat, out);
}
